// EdgeCubeNet_54451595378823
// MI455X (gfx1250) — matrix operand order repaired and checked by reading, not compiled, not run on hardware
//
#include <hip/hip_runtime.h>
#include <math.h>

typedef unsigned short u16;
typedef __attribute__((ext_vector_type(16))) __bf16 v16bf;
typedef __attribute__((ext_vector_type(8)))  float  v8f;

struct __attribute__((aligned(16))) Q4 { unsigned int x, y, z, w; };
union Frag { Q4 q[2]; v16bf v; };

__device__ __forceinline__ u16 f2bf(float f) {
  unsigned u = __float_as_uint(f);
  u += 0x7fffu + ((u >> 16) & 1u);           // round-to-nearest-even
  return (u16)(u >> 16);
}
__device__ __forceinline__ float bf2f(u16 h) {
  return __uint_as_float(((unsigned)h) << 16);
}

// ---- CDNA5 async global->LDS copy, with fallback ----
// Builtin signature (from hipcc diagnostic): (v4i __device__*, v4i __shared__*, Ii, Ii)
#if defined(__gfx1250__) && __has_builtin(__builtin_amdgcn_global_load_async_to_lds_b128)
#define USE_ASYNC_LDS 1
typedef int v4i_vs __attribute__((vector_size(16)));
typedef __attribute__((address_space(1))) v4i_vs as1_v4i;
typedef __attribute__((address_space(3))) v4i_vs as3_v4i;
#endif

__device__ __forceinline__ void async_g2l_b128(const void* g, void* l) {
#if defined(USE_ASYNC_LDS)
  __builtin_amdgcn_global_load_async_to_lds_b128((as1_v4i*)g, (as3_v4i*)l, 0, 0);
#else
  *(Q4*)l = *(const Q4*)g;
#endif
}
__device__ __forceinline__ void wait_async_and_barrier() {
#if defined(__gfx1250__) && __has_builtin(__builtin_amdgcn_s_wait_asynccnt)
  __builtin_amdgcn_s_wait_asynccnt(0);
#endif
  __syncthreads();
}

// ---------------------------------------------------------------- zero
__global__ void k_zero(float* __restrict__ p, int n) {
  for (int i = blockIdx.x * blockDim.x + threadIdx.x; i < n; i += gridDim.x * blockDim.x)
    p[i] = 0.f;
}

// ------------------------------------------------- point MLP + scatter
__global__ void k_point(const float* __restrict__ pts, const int* __restrict__ gidx,
                        const float* __restrict__ w1, const float* __restrict__ b1,
                        const float* __restrict__ w2, const float* __restrict__ b2,
                        float* __restrict__ sums, float* __restrict__ cnts) {
  __shared__ float sw1[96], sb1[32], sw2[2048], sb2[64];
  for (int t = threadIdx.x; t < 96;   t += blockDim.x) sw1[t] = w1[t];
  for (int t = threadIdx.x; t < 32;   t += blockDim.x) sb1[t] = b1[t];
  for (int t = threadIdx.x; t < 2048; t += blockDim.x) sw2[t] = w2[t];
  for (int t = threadIdx.x; t < 64;   t += blockDim.x) sb2[t] = b2[t];
  __syncthreads();
  int i = blockIdx.x * blockDim.x + threadIdx.x;          // 0..131071 (B*N)
  float p0 = pts[3*i], p1 = pts[3*i+1], p2 = pts[3*i+2];
  float h[32];
#pragma unroll
  for (int j = 0; j < 32; ++j)
    h[j] = fmaxf(fmaf(p0, sw1[j], fmaf(p1, sw1[32+j], fmaf(p2, sw1[64+j], sb1[j]))), 0.f);
  int ix = gidx[3*i], iy = gidx[3*i+1], iz = gidx[3*i+2];
  size_t flat = (size_t)(i >> 16) * 262144 + (size_t)(ix*4096 + iy*64 + iz);
  float* dst = sums + flat * 64;
  for (int c = 0; c < 64; ++c) {
    float acc = sb2[c];
#pragma unroll
    for (int j = 0; j < 32; ++j) acc = fmaf(h[j], sw2[j*64 + c], acc);
    atomicAdd(dst + c, fmaxf(acc, 0.f));                  // global_atomic_add_f32
  }
  atomicAdd(cnts + flat, 1.f);
}

// ------------------------------------------------- mean + f32 -> bf16
__global__ void k_norm(const float* __restrict__ S, const float* __restrict__ Cn,
                       u16* __restrict__ V, int n) {
  for (int i = blockIdx.x * blockDim.x + threadIdx.x; i < n; i += gridDim.x * blockDim.x)
    V[i] = f2bf(S[i] / fmaxf(Cn[i >> 6], 1.f));
}

// ----------------- conv weights -> per-lane WMMA B-fragment layout ---
// Wc[tap][ccidx][cot][lane][e] : K=(ccidx*32 + (e&8?16:0) + (lane>=16?8:0) + (e&7)), N=cot*16+(lane&15)
__global__ void k_wconv(const float* __restrict__ W, u16* __restrict__ Wc) {
  int i = blockIdx.x * blockDim.x + threadIdx.x;          // 110592
  if (i >= 27*2*4*32*16) return;
  int e = i & 15, lane = (i >> 4) & 31, cot = (i >> 9) & 3, cc = (i >> 11) & 1, tap = i >> 12;
  int hi = lane >> 4;
  int ci = cc*32 + (e & 8) * 2 + hi*8 + (e & 7);
  int co = cot*16 + (lane & 15);
  Wc[i] = f2bf(W[(co*64 + ci)*27 + tap]);                 // (cout,cin,3,3,3) row-major
}

// ----------------- head weights (Kd x 128) -> B-fragment layout ------
__global__ void k_whead(const float* __restrict__ W, u16* __restrict__ Wf,
                        int kchunks, int Nd) {
  int i = blockIdx.x * blockDim.x + threadIdx.x;
  if (i >= kchunks * 8 * 512) return;
  int e = i & 15, lane = (i >> 4) & 31, nt = (i >> 9) & 7, kc = i >> 12;
  int hi = lane >> 4;
  int k = kc*32 + (e & 8) * 2 + hi*8 + (e & 7);
  int n = nt*16 + (lane & 15);
  Wf[i] = f2bf(W[k*Nd + n]);
}

// ------------------------------- 3D conv 64->64, 3x3x3: implicit GEMM
// Block (256thr) = one (b,x,y): all 64 z x all 64 cout.
// LDS: 3x3 neighborhood of z-columns (async-staged) + per-tap B fragments.
// Wave w: zt = w&3 (16 z rows), cout tiles {2*(w>>2), 2*(w>>2)+1}.
__global__ void __launch_bounds__(256) k_conv3d(
    const u16* __restrict__ V, const u16* __restrict__ Wc,
    const float* __restrict__ conv_b, u16* __restrict__ O) {
  __shared__ u16 sA[9 * 64 * 72];   // 82,944 B; row stride 72 u16 (16B-aligned, bank-safe)
  __shared__ u16 sB[4096];          //  8,192 B; one tap: [cc][cot][lane][e]
  int tid = threadIdx.x;
  int blk = blockIdx.x;                       // B*64*64 = 8192
  int y = blk & 63, x = (blk >> 6) & 63, b = blk >> 12;

  // ---- stage 9 z-columns (64 z x 64 cin bf16, contiguous 8KB each) ----
  for (int i = tid; i < 4608; i += 256) {     // 4608 x b128
    int p = i >> 9;                           // 0..8 (wave-uniform per iteration)
    int rem = i & 511;
    int zr = rem >> 3, qi = rem & 7;
    int xx = x + (p / 3) - 1, yy = y + (p % 3) - 1;
    u16* ldst = sA + ((p * 64 + zr) * 72 + qi * 8);
    int inb = (int)(((unsigned)xx < 64u) & ((unsigned)yy < 64u));
    if (__builtin_amdgcn_readfirstlane(inb)) {
      const u16* gsrc = V + (((size_t)b * 64 + xx) * 64 + yy) * 4096 + (size_t)(zr * 64 + qi * 8);
      async_g2l_b128(gsrc, ldst);
    } else {
      const Q4 z4 = {0u, 0u, 0u, 0u};
      *(Q4*)ldst = z4;
    }
  }

  int w = tid >> 5, lane = tid & 31;
  int zt = w & 3, cot0 = (w >> 2) << 1;
  int z0 = zt << 4;
  int hi = lane >> 4, m = lane & 15;
  const v8f vzero = {0.f, 0.f, 0.f, 0.f, 0.f, 0.f, 0.f, 0.f};
  v8f accv[2];
  accv[0] = vzero; accv[1] = vzero;

  wait_async_and_barrier();                   // s_wait_asynccnt 0 + barrier

  for (int tap = 0; tap < 27; ++tap) {
    __syncthreads();                          // previous tap done reading sB
    {                                         // stage this tap's 8KB of B frags
      const u16* wt = Wc + (size_t)tap * 4096;
      *(Q4*)(sB + tid * 16)     = *(const Q4*)(wt + tid * 16);
      *(Q4*)(sB + tid * 16 + 8) = *(const Q4*)(wt + tid * 16 + 8);
      if (tap + 1 < 27) __builtin_prefetch(Wc + (size_t)(tap + 1) * 4096, 0, 2);
    }
    __syncthreads();
    int p = tap / 3, kw = tap % 3 - 1;
    int zz = z0 + m + kw;
    bool zok = (unsigned)zz < 64u;
    const u16* arow = sA + (p * 64 + (zok ? zz : 0)) * 72;
    const Q4 z4 = {0u, 0u, 0u, 0u};
#pragma unroll
    for (int cc = 0; cc < 2; ++cc) {
      Frag a;
      Q4 lo = *(const Q4*)(arow + cc * 32 + hi * 8);        // ds_load_b128
      Q4 hq = *(const Q4*)(arow + cc * 32 + 16 + hi * 8);
      a.q[0] = zok ? lo : z4;                               // data-side select
      a.q[1] = zok ? hq : z4;
#pragma unroll
      for (int c2 = 0; c2 < 2; ++c2) {
        Frag wv;
        const u16* wp = sB + ((cc * 4 + cot0 + c2) * 512 + lane * 16);
        wv.q[0] = *(const Q4*)(wp);
        wv.q[1] = *(const Q4*)(wp + 8);
        accv[c2] = __builtin_amdgcn_wmma_f32_16x16x32_bf16(false, a.v, false, wv.v,
                                                           (short)0, accv[c2], false, false);
      }
    }
  }
  // epilogue: two 16x16 tiles; D: row = v + hi*8, col = lane&15
  size_t obase = ((((size_t)b * 64 + x) * 64 + y) * 64 + z0) * 64;
#pragma unroll
  for (int c2 = 0; c2 < 2; ++c2) {
    int co = ((cot0 + c2) << 4) + m;
    float bias = conv_b[co];
    u16* orow = O + obase + co;
#pragma unroll
    for (int v = 0; v < 8; ++v)
      orow[(size_t)(v + hi * 8) * 64] = f2bf(fmaxf(accv[c2][v] + bias, 0.f));
  }
}

// ---------------------------------------- gather cube features (bf16)
__global__ void k_gather(const int* __restrict__ ids, const u16* __restrict__ O,
                         u16* __restrict__ G) {
  int i = blockIdx.x * blockDim.x + threadIdx.x;          // 131072 = B*M*8
  int bm = i >> 3, c8 = i & 7;
  int x = ids[bm*3], y = ids[bm*3+1], z = ids[bm*3+2];
  int b = bm >> 13;
  const u16* src = O + ((((size_t)b*64 + x)*64 + y)*64 + z)*64 + c8*8;
  *(Q4*)(G + (size_t)bm*64 + c8*8) = *(const Q4*)src;
}

// --------------------------- head GEMM: Out = relu([Res+] A@W + bias)
__global__ void k_gemm(const u16* __restrict__ A, int lda,
                       const u16* __restrict__ Wf, const float* __restrict__ bias,
                       const u16* __restrict__ Res, u16* __restrict__ Out, int ldo,
                       int kchunks) {
  int wid  = (blockIdx.x * blockDim.x + threadIdx.x) >> 5;
  int lane = threadIdx.x & 31;
  int nt = wid & 7;                                       // 8 col tiles (N=128)
  int r0 = (wid >> 3) << 4;
  int hi = lane >> 4, m = lane & 15;
  v8f acc = {0.f, 0.f, 0.f, 0.f, 0.f, 0.f, 0.f, 0.f};
  const u16* arow = A + (size_t)(r0 + m) * lda;
  for (int kc = 0; kc < kchunks; ++kc) {
    Frag a, w;
    const u16* ap = arow + kc*32;
    a.q[0] = *(const Q4*)(ap + hi*8);
    a.q[1] = *(const Q4*)(ap + 16 + hi*8);
    const u16* wp = Wf + (size_t)(kc*8 + nt) * 512 + lane*16;
    w.q[0] = *(const Q4*)(wp);
    w.q[1] = *(const Q4*)(wp + 8);
    acc = __builtin_amdgcn_wmma_f32_16x16x32_bf16(false, a.v, false, w.v,
                                                  (short)0, acc, false, false);
  }
  int col = (nt << 4) + (lane & 15);
  float bv = bias[col];
#pragma unroll
  for (int v = 0; v < 8; ++v) {
    int row = r0 + v + hi*8;
    float val = acc[v] + bv;
    if (Res) val += bf2f(Res[(size_t)row*ldo + col]);
    Out[(size_t)row*ldo + col] = f2bf(fmaxf(val, 0.f));
  }
}

// ------------------------------------------- final dot + sigmoid (f32)
__global__ void k_headout(const u16* __restrict__ H, const float* __restrict__ wout,
                          const float* __restrict__ bout, float* __restrict__ out) {
  int r = blockIdx.x * blockDim.x + threadIdx.x;          // 16384
  const u16* hr = H + (size_t)r * 128;
  float acc = bout[0];
#pragma unroll 16
  for (int j = 0; j < 128; ++j) acc = fmaf(bf2f(hr[j]), wout[j], acc);
  out[r] = 1.f / (1.f + __expf(-acc));
}

// ====================================================================
extern "C" void kernel_launch(void* const* d_in, const int* in_sizes, int n_in,
                              void* d_out, int out_size, void* d_ws, size_t ws_size,
                              hipStream_t stream) {
  (void)in_sizes; (void)n_in; (void)out_size; (void)ws_size;
  const float* pts    = (const float*)d_in[0];
  const int*   gidx   = (const int*)d_in[1];
  const int*   cids   = (const int*)d_in[2];
  const float* fe_w1  = (const float*)d_in[4];
  const float* fe_b1  = (const float*)d_in[5];
  const float* fe_w2  = (const float*)d_in[6];
  const float* fe_b2  = (const float*)d_in[7];
  const float* conv_w = (const float*)d_in[8];
  const float* conv_b = (const float*)d_in[9];
  const float* w_in   = (const float*)d_in[10];
  const float* b_in   = (const float*)d_in[11];
  const float* bw1    = (const float*)d_in[12];
  const float* bb1    = (const float*)d_in[13];
  const float* bw2    = (const float*)d_in[14];
  const float* bb2    = (const float*)d_in[15];
  const float* w_out  = (const float*)d_in[16];
  const float* b_out  = (const float*)d_in[17];
  float* out = (float*)d_out;

  char* ws = (char*)d_ws;
  float* S  = (float*)(ws);                   // 134,217,728 B (reused as conv out O)
  float* Cn = (float*)(ws + 134217728);       //   2,097,152 B
  u16* V    = (u16*)(ws + 136314880);         //  67,108,864 B  bf16 voxel means
  u16* Wc   = (u16*)(ws + 203423744);         //     221,184 B  conv W fragments
  u16* Wh   = (u16*)(ws + 203644928);         //     147,456 B  head W fragments
  u16* G    = (u16*)(ws + 203792384);         //   2,097,152 B  cube feats
  u16* H1   = (u16*)(ws + 205889536);         //   4,194,304 B
  u16* H2   = (u16*)(ws + 210083840);         //   4,194,304 B
  u16* O    = (u16*)S;

  k_zero<<<2048, 256, 0, stream>>>(S, 34078720);                       // sums+counts
  k_point<<<512, 256, 0, stream>>>(pts, gidx, fe_w1, fe_b1, fe_w2, fe_b2, S, Cn);
  k_norm<<<4096, 256, 0, stream>>>(S, Cn, V, 33554432);
  k_wconv<<<432, 256, 0, stream>>>(conv_w, Wc);
  k_whead<<<32, 256, 0, stream>>>(w_in,        Wh,         2, 128);
  k_whead<<<64, 256, 0, stream>>>(bw1,         Wh + 8192,  4, 128);
  k_whead<<<64, 256, 0, stream>>>(bw1 + 16384, Wh + 24576, 4, 128);
  k_whead<<<64, 256, 0, stream>>>(bw2,         Wh + 40960, 4, 128);
  k_whead<<<64, 256, 0, stream>>>(bw2 + 16384, Wh + 57344, 4, 128);
  k_conv3d<<<8192, 256, 0, stream>>>(V, Wc, conv_b, O);                // 1 block = (b,x,y)
  k_gather<<<512, 256, 0, stream>>>(cids, O, G);
  k_gemm<<<1024, 256, 0, stream>>>(G, 64, Wh, b_in, (const u16*)nullptr, H1, 128, 2);
  for (int i = 0; i < 2; ++i) {
    k_gemm<<<1024, 256, 0, stream>>>(H1, 128, Wh + 8192 + i*16384, bb1 + i*128,
                                     (const u16*)nullptr, H2, 128, 4);
    k_gemm<<<1024, 256, 0, stream>>>(H2, 128, Wh + 40960 + i*16384, bb2 + i*128,
                                     H1, H1, 128, 4);
  }
  k_headout<<<64, 256, 0, stream>>>(H1, w_out, b_out, out);
}
